// FedFNNModel_61186104099204
// MI455X (gfx1250) — matrix operand order repaired and checked by reading, not compiled, not run on hardware
//
#include <hip/hip_runtime.h>
#include <hip/hip_bf16.h>

typedef _Float16 h16;
typedef __attribute__((ext_vector_type(16))) _Float16 v16h;
typedef __attribute__((ext_vector_type(8)))  float    v8f;

__device__ inline v8f wmma16(v16h a, v16h b, v8f c) {
  // D = A(16x32 f16) * B(32x16 f16) + C(16x16 f32)
  return __builtin_amdgcn_wmma_f32_16x16x32_f16(false, a, false, b, (short)0, c, false, false);
}

// ---- WMMA fragment loaders (CDNA5 16-bit layouts per CDNA5 ISA) ----
// A 16x32: lane holds row M=lane%16; half=lane/16.
// vgpr v (pair j=2v+q): K = (v<4 ? half*8+2v : 16+half*8+2(v-4)) + q
template<typename SRC>
__device__ inline v16h load_a_frag(const SRC* A, int lda, int row0, int k0) {
  int lane = threadIdx.x & 31;
  int m = lane & 15, hf = lane >> 4;
  const SRC* p = A + (size_t)(row0 + m) * lda + k0;
  v16h a;
#pragma unroll
  for (int j = 0; j < 16; ++j) {
    int v = j >> 1, q = j & 1;
    int k = ((v < 4) ? (hf * 8 + 2 * v) : (16 + hf * 8 + 2 * (v - 4))) + q;
    a[j] = (_Float16)p[k];
  }
  return a;
}

// B 32x16 from row-major [K,N]: lane holds col N=lane%16; K = (v<4 ? half*8+2v : 16+half*8+2(v-4)) + q
template<typename SRC>
__device__ inline v16h load_b_frag(const SRC* Bm, int ldb, int k0, int n0) {
  int lane = threadIdx.x & 31;
  int n = lane & 15, hf = lane >> 4;
  const SRC* p = Bm + (size_t)k0 * ldb + n0 + n;
  v16h b;
#pragma unroll
  for (int j = 0; j < 16; ++j) {
    int v = j >> 1, q = j & 1;
    int k = ((v < 4) ? (hf * 8 + 2 * v) : (16 + hf * 8 + 2 * (v - 4))) + q;
    b[j] = (_Float16)p[(size_t)k * ldb];
  }
  return b;
}

// B 32x16 where B(k,n) = S[(n0+n)*lds_ + k0+k]  (transposed source, for Q*K^T)
__device__ inline v16h load_bT_frag(const h16* S, int lds_, int k0, int n0) {
  int lane = threadIdx.x & 31;
  int n = lane & 15, hf = lane >> 4;
  const h16* p = S + (size_t)(n0 + n) * lds_ + k0;
  v16h b;
#pragma unroll
  for (int j = 0; j < 16; ++j) {
    int v = j >> 1, q = j & 1;
    int k = ((v < 4) ? (hf * 8 + 2 * v) : (16 + hf * 8 + 2 * (v - 4))) + q;
    b[j] = p[k];
  }
  return b;
}

// ---- C = A[M,K] @ B[K,N] + bias[N], one 16x16 tile per wave ----
template<typename AT>
__global__ void gemm_bias_kernel(const AT* __restrict__ A, const float* __restrict__ Bm,
                                 const float* __restrict__ bias, float* __restrict__ C,
                                 int M, int N, int K) {
  int gw = blockIdx.x * (blockDim.x >> 5) + (threadIdx.x >> 5);
  int ntiles = N >> 4;
  int tm = gw / ntiles, tn = gw % ntiles;
  if (tm >= (M >> 4)) return;  // wave-uniform
  v8f acc = {};
  for (int k0 = 0; k0 < K; k0 += 32) {
    v16h a = load_a_frag<AT>(A, K, tm * 16, k0);
    v16h b = load_b_frag<float>(Bm, N, k0, tn * 16);
    acc = wmma16(a, b, acc);
  }
  int lane = threadIdx.x & 31;
  int col = tn * 16 + (lane & 15);
  int r0  = tm * 16 + (lane >> 4) * 8;
  float bb = bias[col];
#pragma unroll
  for (int r = 0; r < 8; ++r)
    C[(size_t)(r0 + r) * N + col] = acc[r] + bb;
}

// ---- GRU recurrent scan: one block per 16-batch slice; rk B-fragments in VGPRs ----
__global__ void __launch_bounds__(256) gru_scan_kernel(
    const float* __restrict__ xw,      // [B*T, 384] = x@k + bias0
    const float* __restrict__ rk,      // [128, 384]
    const float* __restrict__ biasrow, // bias row 1, [384]
    h16* __restrict__ gout, int T) {
  __shared__ float hS[16 * 128];   // f32 state
  __shared__ h16   hH[16 * 128];   // f16 copy for WMMA A-fragments
  __shared__ float inner[16 * 384];
  const int tid = threadIdx.x;
  const int wave = tid >> 5, lane = tid & 31;
  const int b0 = blockIdx.x * 16;
  for (int i = tid; i < 16 * 128; i += 256) { hS[i] = 0.f; hH[i] = (h16)0.f; }
  // Preload recurrent-weight B fragments for this wave's 3 N-tiles (held in VGPRs)
  v16h bf[3][4];
#pragma unroll
  for (int tle = 0; tle < 3; ++tle)
#pragma unroll
    for (int kk = 0; kk < 4; ++kk)
      bf[tle][kk] = load_b_frag<float>(rk, 384, kk * 32, wave * 48 + tle * 16);
  const int ncol = lane & 15, hf = lane >> 4;
  // loop-invariant bias values for this wave's 3 N-tiles
  const float bb0 = biasrow[wave * 48 + ncol];
  const float bb1 = biasrow[wave * 48 + ncol + 16];
  const float bb2 = biasrow[wave * 48 + ncol + 32];
  __syncthreads();
  for (int t = 0; t < T; ++t) {
    // Preload this step's gate inputs into registers so the global loads
    // overlap the recurrent WMMA below (serial critical path).
    float xzv[8], xrv[8], xhv[8];
#pragma unroll
    for (int i = 0; i < 8; ++i) {
      int e = tid + 256 * i;
      int r = e >> 7, u = e & 127;
      const float* xr_ = xw + ((size_t)(b0 + r) * T + t) * 384;
      xzv[i] = xr_[u]; xrv[i] = xr_[128 + u]; xhv[i] = xr_[256 + u];
    }
    // One-step lookahead: prefetch xw for t+1 (24KB slice) into cache.
    if (t + 1 < T && tid < 192) {
      int pr = tid / 12, cl = tid % 12;
      __builtin_prefetch(xw + ((size_t)(b0 + pr) * T + (t + 1)) * 384 + cl * 32, 0, 1);
    }
    v8f a0 = {}, a1 = {}, a2 = {};
#pragma unroll
    for (int kk = 0; kk < 4; ++kk) {
      v16h a = load_a_frag<h16>(hH, 128, 0, kk * 32);
      a0 = wmma16(a, bf[0][kk], a0);
      a1 = wmma16(a, bf[1][kk], a1);
      a2 = wmma16(a, bf[2][kk], a2);
    }
    {
      int c0 = wave * 48 + ncol;
#pragma unroll
      for (int r = 0; r < 8; ++r) {
        int row = hf * 8 + r;
        inner[row * 384 + c0]      = a0[r] + bb0;
        inner[row * 384 + c0 + 16] = a1[r] + bb1;
        inner[row * 384 + c0 + 32] = a2[r] + bb2;
      }
    }
    __syncthreads();
#pragma unroll
    for (int i = 0; i < 8; ++i) {
      int e = tid + 256 * i;
      int r = e >> 7, u = e & 127;
      float rz = inner[r * 384 + u], rr = inner[r * 384 + 128 + u], rh = inner[r * 384 + 256 + u];
      float z  = 1.f / (1.f + expf(-(xzv[i] + rz)));
      float rg = 1.f / (1.f + expf(-(xrv[i] + rr)));
      float hh = tanhf(xhv[i] + rg * rh);
      float hn = z * hS[e] + (1.f - z) * hh;
      hS[e] = hn;
      hH[e] = (h16)hn;
      gout[((size_t)(b0 + r) * T + t) * 128 + u] = (h16)hn;
    }
    __syncthreads();
  }
}

// ---- Attention + T-mean pool: one wave per (batch, 16-query tile) ----
__global__ void attn_pool_kernel(const h16* __restrict__ g2, float* __restrict__ pooled, int T) {
  __shared__ float sc[16 * 512];   // 32KB: full score rows for this q-tile
  __shared__ float rowm[16], rowl[16];
  int b = blockIdx.x >> 5, qt = blockIdx.x & 31;   // T/16 = 32
  const h16* g = g2 + (size_t)b * T * 128;
  int lane = threadIdx.x & 31;
  int m = lane & 15, hf = lane >> 4;
  // Q fragments (reused across all key tiles)
  v16h aq[4];
#pragma unroll
  for (int kk = 0; kk < 4; ++kk) aq[kk] = load_a_frag<h16>(g, 128, qt * 16, kk * 32);
  for (int st = 0; st < 32; ++st) {
    v8f acc = {};
#pragma unroll
    for (int kk = 0; kk < 4; ++kk) {
      v16h bfr = load_bT_frag(g, 128, kk * 32, st * 16);
      acc = wmma16(aq[kk], bfr, acc);
    }
#pragma unroll
    for (int r = 0; r < 8; ++r)
      sc[(hf * 8 + r) * 512 + st * 16 + m] = acc[r];
  }
  __syncthreads();
  // exact softmax stats: lane handles row m, half of the 512 cols
  float mx = -3.0e38f;
#pragma unroll 8
  for (int c = 0; c < 256; ++c) mx = fmaxf(mx, sc[m * 512 + hf * 256 + c]);
  mx = fmaxf(mx, __shfl_xor(mx, 16));
  float se = 0.f;
#pragma unroll 8
  for (int c = 0; c < 256; ++c) se += expf(sc[m * 512 + hf * 256 + c] - mx);
  se += __shfl_xor(se, 16);
  if (hf == 0) { rowm[m] = mx; rowl[m] = se; }
  __syncthreads();
  // O = P @ V with exp on the fly; 8 N-tiles of U=128
  v8f o[8] = {};
  float myMax = rowm[m];
  for (int kc = 0; kc < 16; ++kc) {
    v16h a;
#pragma unroll
    for (int j = 0; j < 16; ++j) {
      int v = j >> 1, q = j & 1;
      int k = ((v < 4) ? (hf * 8 + 2 * v) : (16 + hf * 8 + 2 * (v - 4))) + q;
      a[j] = (_Float16)expf(sc[m * 512 + kc * 32 + k] - myMax);
    }
#pragma unroll
    for (int nt = 0; nt < 8; ++nt) {
      v16h bfr = load_b_frag<h16>(g, 128, kc * 32, nt * 16);
      o[nt] = wmma16(a, bfr, o[nt]);
    }
  }
  // pooled[b,u] += sum_rows(O/l) / T
#pragma unroll
  for (int nt = 0; nt < 8; ++nt) {
    float s8 = 0.f;
#pragma unroll
    for (int r = 0; r < 8; ++r) s8 += o[nt][r] / rowl[hf * 8 + r];
    s8 += __shfl_xor(s8, 16);
    if (hf == 0) atomicAdd(&pooled[(size_t)b * 128 + nt * 16 + m], s8 * (1.0f / 512.0f));
  }
}

// ---- head: LN + fuzzy rules, one 160-thread block per batch row ----
__device__ inline float red_sum160(float v, float* red) {
  int d = threadIdx.x;
  red[d] = v; __syncthreads();
  if (d < 80) red[d] += red[d + 80]; __syncthreads();
  if (d < 40) red[d] += red[d + 40]; __syncthreads();
  if (d < 20) red[d] += red[d + 20]; __syncthreads();
  if (d < 10) red[d] += red[d + 10]; __syncthreads();
  if (d < 5)  red[d] += red[d + 5];  __syncthreads();
  if (d == 0) red[0] += red[1] + red[2] + red[3] + red[4];
  __syncthreads();
  float r = red[0]; __syncthreads();
  return r;
}
__device__ inline float red_max160(float v, float* red) {
  int d = threadIdx.x;
  red[d] = v; __syncthreads();
  if (d < 80) red[d] = fmaxf(red[d], red[d + 80]); __syncthreads();
  if (d < 40) red[d] = fmaxf(red[d], red[d + 40]); __syncthreads();
  if (d < 20) red[d] = fmaxf(red[d], red[d + 20]); __syncthreads();
  if (d < 10) red[d] = fmaxf(red[d], red[d + 10]); __syncthreads();
  if (d < 5)  red[d] = fmaxf(red[d], red[d + 5]);  __syncthreads();
  if (d == 0) red[0] = fmaxf(fmaxf(fmaxf(red[0], red[1]), fmaxf(red[2], red[3])), red[4]);
  __syncthreads();
  float r = red[0]; __syncthreads();
  return r;
}

__global__ void head_kernel(const float* __restrict__ pooled, const float* __restrict__ stat,
                            const float* __restrict__ gamma, const float* __restrict__ beta,
                            const float* __restrict__ fm, const float* __restrict__ fsig,
                            const float* __restrict__ theta, const float* __restrict__ sflag,
                            const float* __restrict__ rw, const float* __restrict__ rb,
                            float* __restrict__ out) {
  __shared__ float red[160];
  int b = blockIdx.x, d = threadIdx.x;
  float fus = (d < 128) ? pooled[b * 128 + d] : stat[b * 32 + (d - 128)];
  float mu  = red_sum160(fus, red) * (1.0f / 160.0f);
  float dev = fus - mu;
  float var = red_sum160(dev * dev, red) * (1.0f / 160.0f);
  float f   = dev * rsqrtf(var + 1e-5f) * gamma[d] + beta[d];
  float fir[10], ga[10];
  for (int r = 0; r < 10; ++r) {
    float sg  = fmaxf(fabsf(fsig[r * 160 + d]), 1e-6f);
    float mem = fmaxf(0.f, 1.f - fabsf(f - fm[r * 160 + d]) / sg);
    float sm  = -10.f * mem;
    float mxv = red_max160(sm, red);
    float e   = expf(sm - mxv);
    float ssum = red_sum160(e, red);
    float w   = e / ssum;
    float fi  = red_sum160(mem * w, red);
    float gg  = red_sum160(f * theta[r * 161 + 1 + d], red);
    if (d == 0) { fir[r] = fi * sflag[r]; ga[r] = theta[r * 161] + gg; }
  }
  if (d == 0) {
    float fs = 0.f;
    for (int r = 0; r < 10; ++r) fs += fir[r];
    float invs = 1.f / (fs + 1e-9f);
    float fo = 0.f;
    for (int r = 0; r < 10; ++r) fo += fir[r] * invs * ga[r];
    out[b] = fo * rw[0] + rb[0];
  }
}

__global__ void zero_kernel(float* p, int n) {
  int i = blockIdx.x * blockDim.x + threadIdx.x;
  if (i < n) p[i] = 0.f;
}

extern "C" void kernel_launch(void* const* d_in, const int* in_sizes, int n_in,
                              void* d_out, int out_size, void* d_ws, size_t ws_size,
                              hipStream_t stream) {
  (void)in_sizes; (void)n_in; (void)out_size; (void)ws_size;
  const float* ts   = (const float*)d_in[0];
  const float* stat = (const float*)d_in[1];
  const float* k1   = (const float*)d_in[2];
  const float* rk1  = (const float*)d_in[3];
  const float* b1   = (const float*)d_in[4];
  const float* k2   = (const float*)d_in[5];
  const float* rk2  = (const float*)d_in[6];
  const float* b2   = (const float*)d_in[7];
  const float* gam  = (const float*)d_in[8];
  const float* bet  = (const float*)d_in[9];
  const float* fm   = (const float*)d_in[10];
  const float* fsig = (const float*)d_in[11];
  const float* th   = (const float*)d_in[12];
  const float* sfl  = (const float*)d_in[13];
  const float* rw   = (const float*)d_in[14];
  const float* rb   = (const float*)d_in[15];
  float* out = (float*)d_out;

  const int Bb = 128, T = 512, I = 64, U = 128, U3 = 384;
  const int M = Bb * T;
  char* ws = (char*)d_ws;
  float* xw     = (float*)ws;                                                   // M*384 f32 (reused)
  h16*   g1h    = (h16*)(ws + (size_t)M * U3 * 4);                              // M*128 f16
  h16*   g2h    = (h16*)(ws + (size_t)M * U3 * 4 + (size_t)M * U * 2);          // M*128 f16
  float* pooled = (float*)(ws + (size_t)M * U3 * 4 + (size_t)M * U * 4);        // B*128 f32

  int tiles = (M / 16) * (U3 / 16);   // 98304 waves, 8 waves/block
  // Layer 1: xw1 = ts @ k1 + b1[0]; then serial scan
  gemm_bias_kernel<float><<<tiles / 8, 256, 0, stream>>>(ts, k1, b1, xw, M, U3, I);
  gru_scan_kernel<<<Bb / 16, 256, 0, stream>>>(xw, rk1, b1 + U3, g1h, T);
  // Layer 2: xw2 = g1 @ k2 + b2[0]; then serial scan
  gemm_bias_kernel<h16><<<tiles / 8, 256, 0, stream>>>(g1h, k2, b2, xw, M, U3, U);
  gru_scan_kernel<<<Bb / 16, 256, 0, stream>>>(xw, rk2, b2 + U3, g2h, T);
  // Attention + mean pool
  zero_kernel<<<(Bb * U + 255) / 256, 256, 0, stream>>>(pooled, Bb * U);
  attn_pool_kernel<<<Bb * (T / 16), 32, 0, stream>>>(g2h, pooled, T);
  // LN + fuzzy head
  head_kernel<<<Bb, 160, 0, stream>>>(pooled, stat, gam, bet, fm, fsig, th, sfl, rw, rb, out);
}
